// NanoGPT_52527450030376
// MI455X (gfx1250) — compile-verified, matrix operand order checked by reading, not run on hardware
//
#include <hip/hip_runtime.h>
#include <hip/hip_bf16.h>

#define VSZ  32000
#define TSEQ 1024
#define ESZ  768
#define NTH  12
#define CSZ  64
#define BSZ  4
#define LNEPS 1e-5f

typedef __attribute__((ext_vector_type(16))) _Float16 v16h;
typedef __attribute__((ext_vector_type(8)))  float    v8f;

// Load an A/B fragment for v_wmma_f32_16x16x32_f16 from a row-major f16 tile.
// rowbase points at element [row_of_this_lane][k0]; per ISA layout tables:
// pair v holds k = (v/4)*16 + half*8 + (v&3)*2 relative to k0.
// Pairs v=0..3 are bytes [0,16) and v=4..7 bytes [32,48) -> two b128 loads.
__device__ __forceinline__ v16h ldfrag(const _Float16* rowbase, unsigned half) {
  v16h f;
#pragma unroll
  for (int v = 0; v < 8; ++v) {
    int k = ((v >> 2) << 4) + ((int)half << 3) + ((v & 3) << 1);
    f[2 * v]     = rowbase[k];
    f[2 * v + 1] = rowbase[k + 1];
  }
  return f;
}

__device__ __forceinline__ v8f wmma16(v16h a, v16h b, v8f c) {
  return __builtin_amdgcn_wmma_f32_16x16x32_f16(false, a, false, b, (short)0, c,
                                                false, false);
}

// CDNA5 async memory->LDS copy (no VGPR data path, tracked by ASYNCcnt).
__device__ __forceinline__ void async_g2l_b128(unsigned lds_off,
                                               const void* gaddr) {
  asm volatile("global_load_async_to_lds_b128 %0, %1, off"
               :
               : "v"(lds_off), "v"(gaddr)
               : "memory");
}
__device__ __forceinline__ void wait_async0() {
  asm volatile("s_wait_asynccnt 0x0" ::: "memory");
}

// ---------------------------------------------------------------- embedding
__global__ __launch_bounds__(192) void k_embed(const int* __restrict__ x,
                                               const float* __restrict__ tok,
                                               const float* __restrict__ pos,
                                               float* __restrict__ h) {
  int row = blockIdx.x;                   // 0..B*T-1
  int t = row % TSEQ;
  int id = x[row];
  int c4 = threadIdx.x;                   // 0..191  (E/4)
  const float4* te = (const float4*)(tok + (size_t)id * ESZ);
  const float4* pe = (const float4*)(pos + (size_t)t * ESZ);
  float4 a = te[c4], b = pe[c4], o;
  o.x = a.x + b.x; o.y = a.y + b.y; o.z = a.z + b.z; o.w = a.w + b.w;
  ((float4*)(h + (size_t)row * ESZ))[c4] = o;
}

// ------------------------------------------------------------- LN1 + Q/K/V
// grid = NT*B*(T/64), block = 128 (4 waves). Wave w handles 16-row tile w.
// Q,K stored row-major f16; V stored transposed (Vt[c][t]) for P*V B-frags.
__global__ __launch_bounds__(128) void k_ln_qkv(
    const float* __restrict__ h, const float* __restrict__ Wk,
    const float* __restrict__ Wq, const float* __restrict__ Wv,
    const float* __restrict__ g1, const float* __restrict__ b1,
    _Float16* __restrict__ Q, _Float16* __restrict__ K,
    _Float16* __restrict__ Vt) {
  __shared__ _Float16 xs[64][CSZ];        // LN output slab
  __shared__ _Float16 wsq[CSZ * CSZ];     // Wq f16
  __shared__ _Float16 wsk[CSZ * CSZ];     // Wk f16
  __shared__ _Float16 wsv[CSZ * CSZ];     // Wv f16
  int g = blockIdx.x;
  int n = g / (BSZ * (TSEQ / 64));
  int rem = g % (BSZ * (TSEQ / 64));
  int b = rem / (TSEQ / 64);
  int row0 = (rem % (TSEQ / 64)) * 64;
  int tid = threadIdx.x;

  const float* Wq0 = Wq + (size_t)n * CSZ * CSZ;
  const float* Wk0 = Wk + (size_t)n * CSZ * CSZ;
  const float* Wv0 = Wv + (size_t)n * CSZ * CSZ;
  for (int i = tid; i < CSZ * CSZ; i += 128) {
    wsq[i] = (_Float16)Wq0[i];
    wsk[i] = (_Float16)Wk0[i];
    wsv[i] = (_Float16)Wv0[i];
  }
  if (tid < 64) {
    const float* hr = h + ((size_t)(b * TSEQ + row0 + tid) * ESZ + n * CSZ);
    float s = 0.f;
    for (int k = 0; k < CSZ; ++k) s += hr[k];
    float mean = s * (1.f / CSZ);
    float vs = 0.f;
    for (int k = 0; k < CSZ; ++k) { float d = hr[k] - mean; vs += d * d; }
    float inv = rsqrtf(vs * (1.f / CSZ) + LNEPS);
    for (int k = 0; k < CSZ; ++k)
      xs[tid][k] = (_Float16)((hr[k] - mean) * inv * g1[n * CSZ + k] +
                              b1[n * CSZ + k]);
  }
  __syncthreads();

  unsigned lane = tid & 31u, lrow = lane & 15u, half = lane >> 4;
  int wv = tid >> 5;                       // row tile
  size_t nb = (size_t)(n * BSZ + b);
  _Float16* Qb = Q + (nb * TSEQ) * CSZ;
  _Float16* Kb = K + (nb * TSEQ) * CSZ;
  _Float16* Vtb = Vt + (nb * CSZ) * TSEQ;

  v16h a0 = ldfrag(&xs[wv * 16 + lrow][0], half);
  v16h a1 = ldfrag(&xs[wv * 16 + lrow][32], half);
#pragma unroll
  for (int nt = 0; nt < 4; ++nt) {
    const _Float16* wqr = &wsq[(nt * 16 + lrow) * CSZ];
    const _Float16* wkr = &wsk[(nt * 16 + lrow) * CSZ];
    const _Float16* wvr = &wsv[(nt * 16 + lrow) * CSZ];
    v8f aq = {}, ak = {}, av = {};
    aq = wmma16(a0, ldfrag(wqr, half), aq);
    aq = wmma16(a1, ldfrag(wqr + 32, half), aq);
    ak = wmma16(a0, ldfrag(wkr, half), ak);
    ak = wmma16(a1, ldfrag(wkr + 32, half), ak);
    av = wmma16(a0, ldfrag(wvr, half), av);
    av = wmma16(a1, ldfrag(wvr + 32, half), av);
#pragma unroll
    for (int r = 0; r < 8; ++r) {
      int trow = row0 + wv * 16 + (int)half * 8 + r;
      int col = nt * 16 + (int)lrow;
      Qb[(size_t)trow * CSZ + col] = (_Float16)aq[r];
      Kb[(size_t)trow * CSZ + col] = (_Float16)ak[r];
      Vtb[(size_t)col * TSEQ + trow] = (_Float16)av[r];
    }
  }
}

// ---------------------------------------------------- causal flash attention
// grid = NT*B*(T/64), block = 128. Each wave owns one 16-row Q tile and runs
// an independent flash loop over 32-key blocks (no cross-wave barriers).
__global__ __launch_bounds__(128) void k_attn(const _Float16* __restrict__ Q,
                                              const _Float16* __restrict__ K,
                                              const _Float16* __restrict__ Vt,
                                              float* __restrict__ h) {
  __shared__ _Float16 P[4][16][32];       // per-wave P staging (C->A layout)
  int g = blockIdx.x;
  int n = g / (BSZ * (TSEQ / 64));
  int rem = g % (BSZ * (TSEQ / 64));
  int b = rem / (TSEQ / 64);
  int row0 = (rem % (TSEQ / 64)) * 64;
  int tid = threadIdx.x;
  int wv = tid >> 5;
  unsigned lane = tid & 31u, lrow = lane & 15u, half = lane >> 4;
  int t0 = row0 + wv * 16;
  size_t nb = (size_t)(n * BSZ + b);
  const _Float16* Qb = Q + nb * TSEQ * CSZ;
  const _Float16* Kb = K + nb * TSEQ * CSZ;
  const _Float16* Vb = Vt + nb * CSZ * TSEQ;

  v16h aq0 = ldfrag(Qb + (t0 + (int)lrow) * CSZ + 0, half);
  v16h aq1 = ldfrag(Qb + (t0 + (int)lrow) * CSZ + 32, half);

  float mrow[8], lsum[8];
  v8f o[4];
  v8f z = {};
#pragma unroll
  for (int i = 0; i < 4; ++i) o[i] = z;
#pragma unroll
  for (int r = 0; r < 8; ++r) { mrow[r] = -1e30f; lsum[r] = 0.f; }

  for (int s0 = 0; s0 <= t0 + 15; s0 += 32) {
    v8f S0 = {}, S1 = {};
    S0 = wmma16(aq0, ldfrag(Kb + (s0 + (int)lrow) * CSZ + 0, half), S0);
    S0 = wmma16(aq1, ldfrag(Kb + (s0 + (int)lrow) * CSZ + 32, half), S0);
    S1 = wmma16(aq0, ldfrag(Kb + (s0 + 16 + (int)lrow) * CSZ + 0, half), S1);
    S1 = wmma16(aq1, ldfrag(Kb + (s0 + 16 + (int)lrow) * CSZ + 32, half), S1);
#pragma unroll
    for (int r = 0; r < 8; ++r) {
      int trow = t0 + (int)half * 8 + r;
      float s0v = S0[r] * 0.125f;
      float s1v = S1[r] * 0.125f;
      if (s0 + (int)lrow > trow)      s0v = -1e30f;
      if (s0 + 16 + (int)lrow > trow) s1v = -1e30f;
      float vmax = fmaxf(s0v, s1v);
#pragma unroll
      for (int off = 8; off >= 1; off >>= 1)
        vmax = fmaxf(vmax, __shfl_xor(vmax, off, 16));
      float mn = fmaxf(mrow[r], vmax);
      float p0 = __expf(s0v - mn), p1 = __expf(s1v - mn);
      float ps = p0 + p1;
#pragma unroll
      for (int off = 8; off >= 1; off >>= 1) ps += __shfl_xor(ps, off, 16);
      float alpha = __expf(mrow[r] - mn);
      lsum[r] = lsum[r] * alpha + ps;
      mrow[r] = mn;
#pragma unroll
      for (int ct = 0; ct < 4; ++ct) o[ct][r] *= alpha;
      P[wv][(int)half * 8 + r][lrow]      = (_Float16)p0;
      P[wv][(int)half * 8 + r][lrow + 16] = (_Float16)p1;
    }
    v16h ap = ldfrag(&P[wv][lrow][0], half);
#pragma unroll
    for (int ct = 0; ct < 4; ++ct) {
      v16h bv = ldfrag(Vb + (ct * 16 + (int)lrow) * TSEQ + s0, half);
      o[ct] = wmma16(ap, bv, o[ct]);
    }
  }
#pragma unroll
  for (int ct = 0; ct < 4; ++ct)
#pragma unroll
    for (int r = 0; r < 8; ++r) {
      int trow = t0 + (int)half * 8 + r;
      int col = n * CSZ + ct * 16 + (int)lrow;
      h[(size_t)(b * TSEQ + trow) * ESZ + col] += o[ct][r] / lsum[r];
    }
}

// ------------------------------------------------------------ LN2 + FF (+res)
__global__ __launch_bounds__(128) void k_ln_ff(float* __restrict__ h,
                                               const float* __restrict__ ffW,
                                               const float* __restrict__ ffb,
                                               const float* __restrict__ g2,
                                               const float* __restrict__ b2) {
  __shared__ _Float16 xs[64][CSZ];
  __shared__ _Float16 wf[CSZ * CSZ];
  int g = blockIdx.x;
  int n = g / (BSZ * (TSEQ / 64));
  int rem = g % (BSZ * (TSEQ / 64));
  int b = rem / (TSEQ / 64);
  int row0 = (rem % (TSEQ / 64)) * 64;
  int tid = threadIdx.x;
  const float* W = ffW + (size_t)n * CSZ * CSZ;
  for (int i = tid; i < CSZ * CSZ; i += 128) wf[i] = (_Float16)W[i];
  if (tid < 64) {
    const float* hr = h + ((size_t)(b * TSEQ + row0 + tid) * ESZ + n * CSZ);
    float s = 0.f;
    for (int k = 0; k < CSZ; ++k) s += hr[k];
    float mean = s * (1.f / CSZ);
    float vs = 0.f;
    for (int k = 0; k < CSZ; ++k) { float d = hr[k] - mean; vs += d * d; }
    float inv = rsqrtf(vs * (1.f / CSZ) + LNEPS);
    for (int k = 0; k < CSZ; ++k)
      xs[tid][k] = (_Float16)((hr[k] - mean) * inv * g2[n * CSZ + k] +
                              b2[n * CSZ + k]);
  }
  __syncthreads();
  unsigned lane = tid & 31u, lrow = lane & 15u, half = lane >> 4;
  int wv = tid >> 5;
  v16h a0 = ldfrag(&xs[wv * 16 + lrow][0], half);
  v16h a1 = ldfrag(&xs[wv * 16 + lrow][32], half);
#pragma unroll
  for (int nt = 0; nt < 4; ++nt) {
    const _Float16* wr = &wf[(nt * 16 + lrow) * CSZ];
    v8f acc = {};
    acc = wmma16(a0, ldfrag(wr, half), acc);
    acc = wmma16(a1, ldfrag(wr + 32, half), acc);
#pragma unroll
    for (int r = 0; r < 8; ++r) {
      int trow = row0 + wv * 16 + (int)half * 8 + r;
      int col = nt * 16 + (int)lrow;
      size_t idx = (size_t)(b * TSEQ + trow) * ESZ + n * CSZ + col;
      h[idx] += acc[r] + ffb[n * CSZ + col];
    }
  }
}

// ----------------------------------------------------------- f32 -> f16 copy
__global__ void k_cvt16(const float* __restrict__ src,
                        _Float16* __restrict__ dst, int nElem) {
  int i = blockIdx.x * 256 + threadIdx.x;
  if (i < nElem) dst[i] = (_Float16)src[i];
}

// ------------------------------------------------------------------ LM head
// grid = (V/256, BT/32), block = 256 (8 waves). Each wave owns 2 M-tiles x
// 2 N-tiles (wave w -> N-tiles 2w, 2w+1), so every B-fragment fetched from
// L2-resident w16 feeds TWO WMMAs. A slab (32x768 f16, 48 KB) staged in LDS
// via the CDNA5 async global->LDS path (ASYNCcnt), with speculative prefetch
// of the next B k-slice.
__global__ __launch_bounds__(256) void k_lmhead(const _Float16* __restrict__ h16,
                                                const _Float16* __restrict__ w16,
                                                const float* __restrict__ lmb,
                                                float* __restrict__ out) {
  __shared__ _Float16 As[32][ESZ];
  int m0 = blockIdx.y * 32;
  int n0 = blockIdx.x * 256;
  int tid = threadIdx.x;
  {
    // 32*768 halfs = 49152 B = 3072 x b128 chunks; 12 per thread.
    const char* src = (const char*)(h16 + (size_t)m0 * ESZ);
    unsigned lbase = (unsigned)(unsigned long long)(&As[0][0]);
#pragma unroll
    for (int j = 0; j < 12; ++j) {
      int i = tid + j * 256;
      async_g2l_b128(lbase + (unsigned)i * 16u, src + (size_t)i * 16u);
    }
    wait_async0();
  }
  __syncthreads();
  unsigned lane = tid & 31u, lrow = lane & 15u, half = lane >> 4;
  int wv = tid >> 5;
  int nt0 = wv * 2;                       // two adjacent N tiles per wave
  v8f acc00, acc01, acc10, acc11;         // [mt][jn]
  v8f z = {};
  acc00 = z; acc01 = z; acc10 = z; acc11 = z;

  const _Float16* b0p = w16 + (size_t)(n0 + nt0 * 16 + (int)lrow) * ESZ;
  const _Float16* b1p = w16 + (size_t)(n0 + (nt0 + 1) * 16 + (int)lrow) * ESZ;

  for (int k0 = 0; k0 < ESZ; k0 += 32) {
    v16h a0 = ldfrag(&As[lrow][k0], half);        // M-tile 0
    v16h a1 = ldfrag(&As[16 + lrow][k0], half);   // M-tile 1
    if (k0 + 32 < ESZ) {
      __builtin_prefetch(b0p + k0 + 32, 0, 1);
      __builtin_prefetch(b1p + k0 + 32, 0, 1);
    }
    v16h bb0 = ldfrag(b0p + k0, half);
    acc00 = wmma16(a0, bb0, acc00);
    acc10 = wmma16(a1, bb0, acc10);
    v16h bb1 = ldfrag(b1p + k0, half);
    acc01 = wmma16(a0, bb1, acc01);
    acc11 = wmma16(a1, bb1, acc11);
  }

#pragma unroll
  for (int r = 0; r < 8; ++r) {
    int row0r = m0 + (int)half * 8 + r;           // mt = 0
    int row1r = row0r + 16;                       // mt = 1
    int col0 = n0 + nt0 * 16 + (int)lrow;
    int col1 = col0 + 16;
    out[(size_t)row0r * VSZ + col0] = acc00[r] + lmb[col0];
    out[(size_t)row0r * VSZ + col1] = acc01[r] + lmb[col1];
    out[(size_t)row1r * VSZ + col0] = acc10[r] + lmb[col0];
    out[(size_t)row1r * VSZ + col1] = acc11[r] + lmb[col1];
  }
}

// ------------------------------------------------------------------ launcher
extern "C" void kernel_launch(void* const* d_in, const int* in_sizes, int n_in,
                              void* d_out, int out_size, void* d_ws,
                              size_t ws_size, hipStream_t stream) {
  const int*   x   = (const int*)d_in[0];
  const float* tok = (const float*)d_in[1];
  const float* pos = (const float*)d_in[2];
  const float* Wk  = (const float*)d_in[3];
  const float* Wq  = (const float*)d_in[4];
  const float* Wv  = (const float*)d_in[5];
  const float* g1  = (const float*)d_in[6];
  const float* b1  = (const float*)d_in[7];
  const float* ffW = (const float*)d_in[8];
  const float* ffb = (const float*)d_in[9];
  const float* g2  = (const float*)d_in[10];
  const float* b2  = (const float*)d_in[11];
  const float* lmW = (const float*)d_in[12];
  const float* lmb = (const float*)d_in[13];
  float* out = (float*)d_out;

  char* ws = (char*)d_ws;
  // ws layout (bytes): h fp32 | Q f16 | K f16 | Vt f16 | h16 | lmW16  (~87 MB)
  float*    h   = (float*)(ws);                     // 12,582,912
  _Float16* Qh  = (_Float16*)(ws + 12582912);       //  6,291,456
  _Float16* Kh  = (_Float16*)(ws + 18874368);       //  6,291,456
  _Float16* Vth = (_Float16*)(ws + 25165824);       //  6,291,456
  _Float16* h16 = (_Float16*)(ws + 31457280);       //  6,291,456
  _Float16* w16 = (_Float16*)(ws + 37748736);       // 49,152,000

  k_embed<<<BSZ * TSEQ, ESZ / 4, 0, stream>>>(x, tok, pos, h);

  int nbgrid = NTH * BSZ * (TSEQ / 64);             // 768
  k_ln_qkv<<<nbgrid, 128, 0, stream>>>(h, Wk, Wq, Wv, g1, b1, Qh, Kh, Vth);
  k_attn<<<nbgrid, 128, 0, stream>>>(Qh, Kh, Vth, h);
  k_ln_ff<<<nbgrid, 128, 0, stream>>>(h, ffW, ffb, g2, b2);

  int nh = BSZ * TSEQ * ESZ;
  k_cvt16<<<(nh + 255) / 256, 256, 0, stream>>>(h, h16, nh);
  int nw = VSZ * ESZ;
  k_cvt16<<<(nw + 255) / 256, 256, 0, stream>>>(lmW, w16, nw);

  dim3 grid5(VSZ / 256, (BSZ * TSEQ) / 32);         // 125 x 128
  k_lmhead<<<grid5, 256, 0, stream>>>(h16, w16, lmb, out);
}
